// SymmetricLinear_51264729645540
// MI455X (gfx1250) — matrix operand order repaired and checked by reading, not compiled, not run on hardware
//
#include <hip/hip_runtime.h>
#include <stdint.h>

typedef __attribute__((ext_vector_type(16))) _Float16 v16h;
typedef __attribute__((ext_vector_type(4)))  _Float16 v4h;
typedef __attribute__((ext_vector_type(2)))  _Float16 v2h;
typedef __attribute__((ext_vector_type(8)))  float    v8f;

namespace {
constexpr int M_TOT = 8192;     // B*S
constexpr int N_TOT = 16384;    // DOUT
constexpr int K_TOT = 4096;     // DIN
constexpr int BM = 128;
constexpr int BN = 256;
constexpr int BK = 32;          // one 16x16x32 WMMA K-step
constexpr int LDSS = 40;        // padded LDS row stride (80 B) -> conflict-free b128 access
constexpr int NSTEPS = K_TOT / BK;
}

union AFrag { v16h v; float4 q[2]; };

// Exact int8 -> fp16 for 4 packed bytes, 5 VALU ops:
// u = b + 128 (xor 0x80); as_half(0x6400 | u) = 1024 + u (exact); subtract 1152 -> b.
__device__ __forceinline__ void cvt4_i8_f16(uint32_t word, uint32_t& d0, uint32_t& d1) {
  const uint32_t ub = word ^ 0x80808080u;
  const uint32_t lo = __builtin_amdgcn_perm(0x64646464u, ub, 0x04010400u); // {0x64,u1,0x64,u0}
  const uint32_t hi = __builtin_amdgcn_perm(0x64646464u, ub, 0x04030402u); // {0x64,u3,0x64,u2}
  const v2h off = { (_Float16)(-1152.0f), (_Float16)(-1152.0f) };
  d0 = __builtin_bit_cast(uint32_t, __builtin_bit_cast(v2h, lo) + off);    // v_pk_add_f16
  d1 = __builtin_bit_cast(uint32_t, __builtin_bit_cast(v2h, hi) + off);
}

__global__ __launch_bounds__(256)
void symlin_wmma_kernel(const float* __restrict__ x,
                        const int8_t* __restrict__ w,
                        const float* __restrict__ scale_p,
                        const float* __restrict__ bias,
                        float* __restrict__ out)
{
  // Double-buffered staging: 20 KB (A) + 40 KB (B) = 60 KB static LDS.
  __shared__ alignas(16) _Float16 sA[2][BM * LDSS];
  __shared__ alignas(16) _Float16 sB[2][BN * LDSS];

  const int tid  = threadIdx.x;
  const int lane = tid & 31;
  const int wave = tid >> 5;
  const int wm   = wave >> 2;    // 0..1 (M)
  const int wn   = wave & 3;     // 0..3 (N)

  const int n0 = blockIdx.x * BN;    // N fastest-varying -> x stripes reused from 192MB L2
  const int m0 = blockIdx.y * BM;

  v8f acc[4][4];
#pragma unroll
  for (int mi = 0; mi < 4; ++mi)
#pragma unroll
    for (int ni = 0; ni < 4; ++ni)
      acc[mi][ni] = (v8f)0.0f;

  // fragment addressing per ISA wave32 layouts
  const int fm = lane & 15;
  const int ka = (lane < 16) ? 0 : 8;    // A: halves {ka..ka+7, ka+16..ka+23}
  const int kb = (lane < 16) ? 0 : 8;    // B: halves {kb..kb+7, kb+16..kb+23}

  // staging-to-register state (lives across compute for double buffering)
  float4 aReg[4];
  int4   bReg[2];
  const int brow = tid;                  // one B row (32 bytes) per thread

  auto gload = [&](int k0) {
#pragma unroll
    for (int q = 0; q < 4; ++q) {
      const int c   = tid + (q << 8);    // 0..1023
      const int row = c >> 3;            // 0..127
      const int jc  = (c & 7) << 2;      // 0..28
      aReg[q] = *(const float4*)(x + (size_t)(m0 + row) * K_TOT + (k0 + jc));
    }
    const int8_t* wp = w + (size_t)(n0 + brow) * K_TOT + k0;
    bReg[0] = *(const int4*)(wp);
    bReg[1] = *(const int4*)(wp + 16);
  };

  auto stage = [&](int buf) {
#pragma unroll
    for (int q = 0; q < 4; ++q) {
      const int c   = tid + (q << 8);
      const int row = c >> 3;
      const int jc  = (c & 7) << 2;
      v4h t;
      t[0] = (_Float16)aReg[q].x; t[1] = (_Float16)aReg[q].y;   // v_cvt_pk_f16_f32 x2
      t[2] = (_Float16)aReg[q].z; t[3] = (_Float16)aReg[q].w;
      *(v4h*)(&sA[buf][row * LDSS + jc]) = t;                    // ds_store_b64
    }
    _Float16* dstb = &sB[buf][brow * LDSS];
#pragma unroll
    for (int p = 0; p < 2; ++p) {
      uint32_t d0, d1, d2, d3, d4, d5, d6, d7;
      cvt4_i8_f16((uint32_t)bReg[p].x, d0, d1);
      cvt4_i8_f16((uint32_t)bReg[p].y, d2, d3);
      cvt4_i8_f16((uint32_t)bReg[p].z, d4, d5);
      cvt4_i8_f16((uint32_t)bReg[p].w, d6, d7);
      uint4 lo4 = {d0, d1, d2, d3};
      uint4 hi4 = {d4, d5, d6, d7};
      *(uint4*)(dstb + p * 16)     = lo4;                        // ds_store_b128
      *(uint4*)(dstb + p * 16 + 8) = hi4;
    }
  };

  auto compute = [&](int buf) {
    AFrag aF[4];
#pragma unroll
    for (int mi = 0; mi < 4; ++mi) {
      const _Float16* s = &sA[buf][(wm * 64 + mi * 16 + fm) * LDSS + ka];
      aF[mi].q[0] = *(const float4*)(s);        // K ka..ka+7
      aF[mi].q[1] = *(const float4*)(s + 16);   // K ka+16..ka+23
    }
#pragma unroll
    for (int ni = 0; ni < 4; ++ni) {
      AFrag bF;
      const _Float16* s = &sB[buf][(wn * 64 + ni * 16 + fm) * LDSS + kb];
      bF.q[0] = *(const float4*)(s);
      bF.q[1] = *(const float4*)(s + 16);
#pragma unroll
      for (int mi = 0; mi < 4; ++mi) {
        acc[mi][ni] = __builtin_amdgcn_wmma_f32_16x16x32_f16(
            false, aF[mi].v, false, bF.v, (short)0, acc[mi][ni], false, false);
      }
    }
  };

  // ---- software pipeline: one barrier per K-step, loads overlap 16 WMMAs
  gload(0);
  stage(0);
  __syncthreads();
  for (int step = 0; step < NSTEPS; ++step) {
    if (step + 1 < NSTEPS) gload((step + 1) * BK);   // issue early; consumed in stage()
    compute(step & 1);
    if (step + 1 < NSTEPS) stage((step + 1) & 1);    // writes the other buffer
    __syncthreads();
  }

  // ---- epilogue: scale + bias, non-temporal stores keep the 536MB output out of L2
  const float scale = scale_p[0];
  const int col0 = n0 + wn * 64 + fm;
  const int rAdd = (lane < 16) ? 0 : 8;              // C/D layout: lanes 16-31 hold M+8
#pragma unroll
  for (int ni = 0; ni < 4; ++ni) {
    const int col = col0 + ni * 16;
    const float bv = bias[col];
#pragma unroll
    for (int mi = 0; mi < 4; ++mi) {
      const int rowb = m0 + wm * 64 + mi * 16 + rAdd;
#pragma unroll
      for (int r = 0; r < 8; ++r) {
        const float val = acc[mi][ni][r] * scale + bv;
        __builtin_nontemporal_store(val, out + (size_t)(rowb + r) * N_TOT + col);
      }
    }
  }
}

extern "C" void kernel_launch(void* const* d_in, const int* in_sizes, int n_in,
                              void* d_out, int out_size, void* d_ws, size_t ws_size,
                              hipStream_t stream) {
  (void)in_sizes; (void)n_in; (void)out_size; (void)d_ws; (void)ws_size;
  const float*   x     = (const float*)d_in[0];
  const int8_t*  w8    = (const int8_t*)d_in[1];
  const float*   scale = (const float*)d_in[2];
  const float*   bias  = (const float*)d_in[3];
  float*         out   = (float*)d_out;

  dim3 grid(N_TOT / BN, M_TOT / BM);   // 64 x 64; N varies fastest for L2 reuse of x stripes
  dim3 block(256);                     // 8 waves (wave32)
  symlin_wmma_kernel<<<grid, block, 0, stream>>>(x, w8, scale, bias, out);
}
